// NGPNeuralField_84963043049632
// MI455X (gfx1250) — matrix operand order repaired and checked by reading, not compiled, not run on hardware
//
#include <hip/hip_runtime.h>
#include <stdint.h>

typedef __attribute__((ext_vector_type(16))) _Float16 v16h;
typedef __attribute__((ext_vector_type(8)))  _Float16 v8h;
typedef __attribute__((ext_vector_type(8)))  float    v8f;

#define N_LEVELS 16
#define LOG2_T   19
#define TSIZE    (1u << LOG2_T)
#define TMASK    (TSIZE - 1u)
#define NPTS_BLK 256
#define FSTR     40   // feats LDS row stride (halves): 80B rows, conflict-free b128 reads
#define H1STR    72   // h1 LDS row stride (halves): 144B rows, conflict-free b128 reads

__global__ __launch_bounds__(256)
void ngp_fused_kernel(const float* __restrict__ pos,
                      const float* __restrict__ tables,
                      const float* __restrict__ w1,
                      const float* __restrict__ w2,
                      const float* __restrict__ w3,
                      float* __restrict__ out,
                      int npts)
{
    __shared__ __align__(16) _Float16 sFeats[NPTS_BLK * FSTR];   // 20 KB
    __shared__ __align__(16) _Float16 sW1T[64 * 32];             // 4 KB  [N][K]
    __shared__ __align__(16) _Float16 sW2T[64 * 64];             // 8 KB  [N][K]
    __shared__ __align__(16) _Float16 sH1[8 * 16 * H1STR];       // 18 KB per-wave scratch
    __shared__ float sSel[NPTS_BLK];

    const int tid = threadIdx.x;
    const int blk = blockIdx.x;

    // ---------------- stage weights: f32 -> f16, transposed to [N][K] ----------------
    for (int i = tid; i < 32 * 64; i += 256) {
        int k = i >> 6, n = i & 63;                 // w1 is [K=32][N=64] row-major
        sW1T[n * 32 + k] = (_Float16)w1[i];
    }
    for (int i = tid; i < 64 * 64; i += 256) {
        int k = i >> 6, n = i & 63;                 // w2 is [K=64][N=64] row-major
        sW2T[n * 64 + k] = (_Float16)w2[i];
    }

    // ---------------- phase 1: hash-grid encoding, one point per thread ----------------
    {
        const int gp = blk * NPTS_BLK + tid;
        _Float16* frow = &sFeats[tid * FSTR];
        if (gp < npts) {
            float px = pos[3 * gp + 0];
            float py = pos[3 * gp + 1];
            float pz = pos[3 * gp + 2];
            // AABB [-1,1]^3 -> unit cube
            float x = (px + 1.0f) * 0.5f;
            float y = (py + 1.0f) * 0.5f;
            float z = (pz + 1.0f) * 0.5f;
            bool sel = (x > 0.0f) & (x < 1.0f) & (y > 0.0f) & (y < 1.0f) &
                       (z > 0.0f) & (z < 1.0f);
            sSel[tid] = sel ? 1.0f : 0.0f;
            float xc = fminf(fmaxf(x, 0.0f), 1.0f);
            float yc = fminf(fmaxf(y, 0.0f), 1.0f);
            float zc = fminf(fmaxf(z, 0.0f), 1.0f);

            // floor(16 * 1.5^l) in f32 (exact)
            const float RESL[16] = {16.f, 24.f, 36.f, 54.f, 81.f, 121.f, 182.f, 273.f,
                                    410.f, 615.f, 922.f, 1383.f, 2075.f, 3113.f, 4670.f, 7006.f};
#pragma unroll
            for (int lv = 0; lv < N_LEVELS; ++lv) {
                const float r = RESL[lv];
                float fx = xc * r, fy = yc * r, fz = zc * r;
                float p0x = floorf(fx), p0y = floorf(fy), p0z = floorf(fz);
                float wx = fx - p0x, wy = fy - p0y, wz = fz - p0z;
                uint32_t ix = (uint32_t)p0x, iy = (uint32_t)p0y, iz = (uint32_t)p0z;
                const float2* tb = (const float2*)tables + (size_t)lv * TSIZE;
                float f0 = 0.0f, f1 = 0.0f;
#pragma unroll
                for (int c = 0; c < 8; ++c) {
                    uint32_t ux = ix + ((c >> 2) & 1);
                    uint32_t uy = iy + ((c >> 1) & 1);
                    uint32_t uz = iz + (c & 1);
                    uint32_t h = (ux ^ (uy * 2654435761u) ^ (uz * 805459861u)) & TMASK;
                    float2 e = tb[h];
                    float cw = (((c >> 2) & 1) ? wx : 1.0f - wx) *
                               (((c >> 1) & 1) ? wy : 1.0f - wy) *
                               (((c     ) & 1) ? wz : 1.0f - wz);
                    f0 = fmaf(e.x, cw, f0);
                    f1 = fmaf(e.y, cw, f1);
                }
                union { _Float16 hh[2]; uint32_t u; } pk;
                pk.hh[0] = (_Float16)f0;
                pk.hh[1] = (_Float16)f1;
                *(uint32_t*)(frow + 2 * lv) = pk.u;   // packed 4B DS store
            }
        } else {
            sSel[tid] = 0.0f;
#pragma unroll
            for (int lv = 0; lv < N_LEVELS; ++lv)
                *(uint32_t*)(frow + 2 * lv) = 0u;
        }
    }
    __syncthreads();

    // ---------------- phase 2: fused MLP via WMMA (wave32) ----------------
    const int lane = tid & 31;
    const int wv   = tid >> 5;
    const int lo   = lane & 15;     // B/C column within tile
    const int hi   = lane >> 4;     // K-half selector (A/B), M-half selector (C)

    // Persistent B fragments (B elem j -> K = j + 8*hi + 8*(j>=8), N = lo)
    v16h b1[4];
#pragma unroll
    for (int t = 0; t < 4; ++t)
        b1[t] = __builtin_shufflevector(*(const v8h*)&sW1T[(t * 16 + lo) * 32 + 8 * hi],
                                        *(const v8h*)&sW1T[(t * 16 + lo) * 32 + 16 + 8 * hi],
                                        0,1,2,3,4,5,6,7,8,9,10,11,12,13,14,15);
    v16h b2[4][2];
#pragma unroll
    for (int t = 0; t < 4; ++t)
#pragma unroll
        for (int kc = 0; kc < 2; ++kc)
            b2[t][kc] = __builtin_shufflevector(*(const v8h*)&sW2T[(t * 16 + lo) * 64 + kc * 32 + 8 * hi],
                                                *(const v8h*)&sW2T[(t * 16 + lo) * 64 + kc * 32 + 16 + 8 * hi],
                                                0,1,2,3,4,5,6,7,8,9,10,11,12,13,14,15);
    float w3v[4];
#pragma unroll
    for (int t = 0; t < 4; ++t) w3v[t] = w3[t * 16 + lo];

    _Float16* hb = &sH1[wv * 16 * H1STR];
    const v8f zero = {};

#pragma unroll
    for (int mt = 0; mt < 2; ++mt) {
        const int rbase = wv * 32 + mt * 16;    // local row base for this 16-point tile

        // A1 fragment: A elem j -> K = j + 8*hi + 8*(j>=8), M = lo (two 16B DS reads)
        const _Float16* fr = &sFeats[(rbase + lo) * FSTR];
        v8h a1r0 = *(const v8h*)(fr + 8 * hi);
        v8h a1r1 = *(const v8h*)(fr + 16 + 8 * hi);
        v16h a1 = __builtin_shufflevector(a1r0, a1r1,
                                          0,1,2,3,4,5,6,7,8,9,10,11,12,13,14,15);

        // GEMM1: [16x32] @ [32x64] -> 4 N-tiles
        v8f c1[4];
#pragma unroll
        for (int t = 0; t < 4; ++t)
            c1[t] = __builtin_amdgcn_wmma_f32_16x16x32_f16(
                        false, a1, false, b1[t], (short)0, zero, false, false);

        // ReLU -> f16 -> LDS (C layout -> row-major [M][K] for the next A load)
#pragma unroll
        for (int t = 0; t < 4; ++t)
#pragma unroll
            for (int m = 0; m < 8; ++m)
                hb[(m + 8 * hi) * H1STR + t * 16 + lo] =
                    (_Float16)fmaxf(c1[t][m], 0.0f);

        // A2 fragments (K = 0..31 and 32..63)
        const _Float16* hr = hb + lo * H1STR;
        v8h q0 = *(const v8h*)(hr + 8 * hi);
        v8h q1 = *(const v8h*)(hr + 16 + 8 * hi);
        v8h q2 = *(const v8h*)(hr + 32 + 8 * hi);
        v8h q3 = *(const v8h*)(hr + 48 + 8 * hi);
        v16h a20 = __builtin_shufflevector(q0, q1, 0,1,2,3,4,5,6,7,8,9,10,11,12,13,14,15);
        v16h a21 = __builtin_shufflevector(q2, q3, 0,1,2,3,4,5,6,7,8,9,10,11,12,13,14,15);

        // GEMM2: [16x64] @ [64x64], K=64 chained through C operand
        v8f c2[4];
#pragma unroll
        for (int t = 0; t < 4; ++t) {
            v8f acc = __builtin_amdgcn_wmma_f32_16x16x32_f16(
                          false, a20, false, b2[t][0], (short)0, zero, false, false);
            c2[t]  = __builtin_amdgcn_wmma_f32_16x16x32_f16(
                          false, a21, false, b2[t][1], (short)0, acc, false, false);
        }

        // Final 64->1 layer: per-lane partial dot with w3, butterfly reduce within
        // each 16-lane half (C layout: lanes 0-15 hold M=0..7, lanes 16-31 hold M=8..15)
#pragma unroll
        for (int m = 0; m < 8; ++m) {
            float s = fmaxf(c2[0][m], 0.0f) * w3v[0]
                    + fmaxf(c2[1][m], 0.0f) * w3v[1]
                    + fmaxf(c2[2][m], 0.0f) * w3v[2]
                    + fmaxf(c2[3][m], 0.0f) * w3v[3];
            s += __shfl_xor(s, 1, 32);
            s += __shfl_xor(s, 2, 32);
            s += __shfl_xor(s, 4, 32);
            s += __shfl_xor(s, 8, 32);
            if (lo == 0) {
                int lrow = rbase + m + 8 * hi;
                int gp = blk * NPTS_BLK + lrow;
                if (gp < npts) {
                    float v = s * sSel[lrow];        // AABB selector mask
                    out[gp] = fmaxf(v, 0.0f);        // final ReLU
                }
            }
        }
    }
}

extern "C" void kernel_launch(void* const* d_in, const int* in_sizes, int n_in,
                              void* d_out, int out_size, void* d_ws, size_t ws_size,
                              hipStream_t stream) {
    const float* pos    = (const float*)d_in[0];   // [N,3] f32
    const float* tables = (const float*)d_in[1];   // [16, 2^19, 2] f32
    const float* w1     = (const float*)d_in[2];   // [32,64] f32
    const float* w2     = (const float*)d_in[3];   // [64,64] f32
    const float* w3     = (const float*)d_in[4];   // [64,1]  f32
    float* out = (float*)d_out;                    // [N,1]   f32

    const int npts = in_sizes[0] / 3;
    const int blocks = (npts + NPTS_BLK - 1) / NPTS_BLK;
    ngp_fused_kernel<<<blocks, 256, 0, stream>>>(pos, tables, w1, w2, w3, out, npts);
}
